// InterferenceDecoder_27230092657280
// MI455X (gfx1250) — compile-verified, matrix operand order checked by reading, not run on hardware
//
#include <hip/hip_runtime.h>

typedef __attribute__((ext_vector_type(16))) __bf16 v16bf;
typedef __attribute__((ext_vector_type(8)))  float  v8f;

#define SEQ   2048
#define EMBED 128
#define VOCAB 50257
#define VPAD  50304        // VOCAB rounded up to 64 (4 waves x 16 cols)
#define MROWS 128          // psi rows staged per block
#define LDSTR 136          // LDS row stride in bf16: 128 + 8 pad -> 4-dword bank rotate/row

// ---------- fp32 -> bf16 (RNE) ----------
__device__ __forceinline__ unsigned short f2bf(float f) {
  unsigned u = __float_as_uint(f);
  u += 0x7FFFu + ((u >> 16) & 1u);
  return (unsigned short)(u >> 16);
}

// dst = bf16(src), rows >= src_rows zero-padded
__global__ void cvt_bf16_pad(const float* __restrict__ src, unsigned short* __restrict__ dst,
                             int src_rows, long total, int cols) {
  long i = (long)blockIdx.x * blockDim.x + threadIdx.x;
  const long stride = (long)gridDim.x * blockDim.x;
  for (; i < total; i += stride) {
    long row = i / cols;
    dst[i] = (row < (long)src_rows) ? f2bf(src[i]) : (unsigned short)0;
  }
}

// dst = bf16(a + sign*b)  (combine done in fp32 before rounding)
__global__ void cvt2_bf16_pad(const float* __restrict__ a, const float* __restrict__ b, float sign,
                              unsigned short* __restrict__ dst, int src_rows, long total, int cols) {
  long i = (long)blockIdx.x * blockDim.x + threadIdx.x;
  const long stride = (long)gridDim.x * blockDim.x;
  for (; i < total; i += stride) {
    long row = i / cols;
    dst[i] = (row < (long)src_rows) ? f2bf(fmaf(sign, b[i], a[i])) : (unsigned short)0;
  }
}

// ---------- fragment helper ----------
union Frag {
  uint4 u[2];
  v16bf bf;
};

// 16 contiguous bf16 (32B) per lane; lane%16 = row(A)/col(B), lane/16 = K half.
__device__ __forceinline__ v16bf load_frag(const unsigned short* p) {
  Frag x;
  x.u[0] = *(const uint4*)(p);
  x.u[1] = *(const uint4*)(p + 8);
  return x.bf;
}

__device__ __forceinline__ v8f wmma_bf16(v16bf a, v16bf b, v8f c) {
  return __builtin_amdgcn_wmma_f32_16x16x32_bf16(false, a, false, b, (short)0, c, false, false);
}

// ---------- main fused kernel ----------
// Karatsuba complex GEMM: P1 = ψr·Pr, P2 = ψi·Pi, P3 = (ψr-ψi)·(Pr+Pi), L = ψr·W
//   Sr = P1+P2 ; Si = P3-P1+P2 ; out = Sr^2 + Si^2 + L + b   (16 WMMA / tile)
__global__ void __launch_bounds__(128)
interference_gemm(const unsigned short* __restrict__ psiR,
                  const unsigned short* __restrict__ psiI,
                  const unsigned short* __restrict__ psiD,   // ψr - ψi
                  const unsigned short* __restrict__ patR,
                  const unsigned short* __restrict__ patI,
                  const unsigned short* __restrict__ patS,   // Pr + Pi
                  const unsigned short* __restrict__ Wb,
                  const float* __restrict__ bias,
                  float* __restrict__ out) {
  extern __shared__ unsigned short lds[];
  unsigned short* ldsR = lds;
  unsigned short* ldsI = lds + MROWS * LDSTR;
  unsigned short* ldsD = lds + 2 * MROWS * LDSTR;

  const int tid   = threadIdx.x;
  const int lane  = tid & 31;
  const int wave  = tid >> 5;
  const int n     = lane & 15;
  const int khalf = lane >> 4;
  const int kofs  = khalf * 16;

  const int v0    = (blockIdx.x * 4 + wave) * 16;
  const int col   = v0 + n;
  const int m0blk = blockIdx.y * MROWS;

  // --- resident B fragments: Pr, Pi, Pr+Pi, W for all four K=32 chunks (128 VGPRs) ---
  v16bf Br[4], Bi[4], Bs[4], Wf[4];
#pragma unroll
  for (int kc = 0; kc < 4; ++kc) {
    const size_t boff = (size_t)(v0 + n) * EMBED + kc * 32 + kofs;
    Br[kc] = load_frag(patR + boff);
    Bi[kc] = load_frag(patI + boff);
    Bs[kc] = load_frag(patS + boff);
    Wf[kc] = load_frag(Wb  + boff);
  }

  // --- cooperative stage of the three ψ arrays into LDS (uint4 granularity) ---
  for (int idx = tid; idx < MROWS * 16; idx += 128) {   // 16 uint4 per 128-elem row
    const int row = idx >> 4, c = idx & 15;
    const size_t g = (size_t)(m0blk + row) * EMBED + c * 8;
    const int l = row * LDSTR + c * 8;
    *(uint4*)(ldsR + l) = *(const uint4*)(psiR + g);
    *(uint4*)(ldsI + l) = *(const uint4*)(psiI + g);
    *(uint4*)(ldsD + l) = *(const uint4*)(psiD + g);
  }
  __syncthreads();

  const float bv = (col < VOCAB) ? bias[col] : 0.0f;

  for (int t = 0; t < MROWS / 16; ++t) {
    const int rbase = (t * 16 + n) * LDSTR;

    v8f P1 = {}, P2 = {}, P3 = {}, L = {};
#pragma unroll
    for (int kc = 0; kc < 4; ++kc) {
      const int o = rbase + kc * 32 + kofs;
      v16bf Ar = load_frag(ldsR + o);   // ds_load_b128 x2 each
      v16bf Ai = load_frag(ldsI + o);
      v16bf Ad = load_frag(ldsD + o);
      P1 = wmma_bf16(Ar, Br[kc], P1);
      P2 = wmma_bf16(Ai, Bi[kc], P2);
      P3 = wmma_bf16(Ad, Bs[kc], P3);
      L  = wmma_bf16(Ar, Wf[kc], L);
    }

    if (col < VOCAB) {
      const int mbase = m0blk + t * 16;
      // 16x16 f32 C/D layout: VGPR r, lanes 0-15 -> M=r, lanes 16-31 -> M=8+r; N=lane%16
#pragma unroll
      for (int r = 0; r < 8; ++r) {
        const int m  = mbase + r + 8 * khalf;
        const float sr = P1[r] + P2[r];
        const float si = P3[r] - P1[r] + P2[r];
        out[(size_t)m * VOCAB + col] = sr * sr + si * si + L[r] + bv;
      }
    }
  }
}

extern "C" void kernel_launch(void* const* d_in, const int* in_sizes, int n_in,
                              void* d_out, int out_size, void* d_ws, size_t ws_size,
                              hipStream_t stream) {
  const float* psi_real      = (const float*)d_in[0];
  const float* psi_imag      = (const float*)d_in[1];
  const float* patterns_real = (const float*)d_in[2];
  const float* patterns_imag = (const float*)d_in[3];
  const float* W             = (const float*)d_in[4];
  const float* b             = (const float*)d_in[5];
  float* out = (float*)d_out;

  // Workspace (bf16): psiR | psiI | psiD | patR | patI | patS | W  (~53 MB)
  const long PSI_N = (long)SEQ * EMBED;
  const long PAT_N = (long)VPAD * EMBED;
  unsigned short* psiRb = (unsigned short*)d_ws;
  unsigned short* psiIb = psiRb + PSI_N;
  unsigned short* psiDb = psiIb + PSI_N;
  unsigned short* patRb = psiDb + PSI_N;
  unsigned short* patIb = patRb + PAT_N;
  unsigned short* patSb = patIb + PAT_N;
  unsigned short* Wbb   = patSb + PAT_N;

  cvt_bf16_pad <<<512,  256, 0, stream>>>(psi_real, psiRb, SEQ, PSI_N, EMBED);
  cvt_bf16_pad <<<512,  256, 0, stream>>>(psi_imag, psiIb, SEQ, PSI_N, EMBED);
  cvt2_bf16_pad<<<512,  256, 0, stream>>>(psi_real, psi_imag, -1.0f, psiDb, SEQ, PSI_N, EMBED);
  cvt_bf16_pad <<<4096, 256, 0, stream>>>(patterns_real, patRb, VOCAB, PAT_N, EMBED);
  cvt_bf16_pad <<<4096, 256, 0, stream>>>(patterns_imag, patIb, VOCAB, PAT_N, EMBED);
  cvt2_bf16_pad<<<4096, 256, 0, stream>>>(patterns_real, patterns_imag, 1.0f, patSb, VOCAB, PAT_N, EMBED);
  cvt_bf16_pad <<<4096, 256, 0, stream>>>(W, Wbb, VOCAB, PAT_N, EMBED);

  const size_t lds_bytes = (size_t)MROWS * LDSTR * 3 * sizeof(unsigned short); // ~102 KB
  dim3 grid(VPAD / 64, SEQ / MROWS);   // 786 x 16 blocks, 4 waves each
  interference_gemm<<<grid, 128, lds_bytes, stream>>>(psiRb, psiIb, psiDb,
                                                      patRb, patIb, patSb, Wbb, b, out);
}
